// TxtNet_V1_88364657148583
// MI455X (gfx1250) — matrix operand order repaired and checked by reading, not compiled, not run on hardware
//
#include <hip/hip_runtime.h>
#include <hip/hip_bf16.h>
#include <math.h>

typedef __bf16 bf16_t;
typedef __attribute__((ext_vector_type(16))) __bf16 v16bf;
typedef __attribute__((ext_vector_type(8)))  __bf16 v8bf;
typedef __attribute__((ext_vector_type(8)))  float   v8f;

#define N_NODES 8192
#define N_EDGES 8192
#define N_NNZ   65536
#define TXT_FEAT 1386
#define KPAD1   1408          // 1386 padded to multiple of 32
#define HIDF    4096
#define CODEF   64

#define BROW    40            // LDS row stride (bf16 elems): 20 words -> r*20 mod 64 all
                              // distinct for r=0..15 => conflict-free b128 LDS reads

// Use the gfx1250 async global->LDS path for B-panel staging when available.
#if defined(__has_builtin)
#if __has_builtin(__builtin_amdgcn_global_load_async_to_lds_b128) && \
    __has_builtin(__builtin_amdgcn_s_wait_asynccnt)
#define USE_ASYNC_LDS 1
#endif
#endif

#if defined(USE_ASYNC_LDS)
__device__ __forceinline__ void async_stage_b128(const bf16_t* g, bf16_t* l) {
  typedef int v4i __attribute__((vector_size(16)));
  typedef __attribute__((address_space(1))) v4i gv4i;
  typedef __attribute__((address_space(3))) v4i lv4i;
  v4i* gg = (v4i*)const_cast<bf16_t*>(g);   // generic bitcast
  v4i* ll = (v4i*)l;                        // generic bitcast
  __builtin_amdgcn_global_load_async_to_lds_b128(
      (gv4i*)gg,                            // addrspace(1) cast
      (lv4i*)ll,                            // addrspace(3) cast
      0, 0);
}
#endif

// ---------------- elementwise / utility kernels ----------------

__global__ void zero_f32_kernel(float* __restrict__ p, int n4) {
  int i = blockIdx.x * blockDim.x + threadIdx.x;
  if (i < n4) ((float4*)p)[i] = make_float4(0.f, 0.f, 0.f, 0.f);
}

// integer degree counting: order-independent, exact
__global__ void degree_count_kernel(const int* __restrict__ nidx,
                                    const int* __restrict__ eidx,
                                    unsigned* __restrict__ Dc,
                                    unsigned* __restrict__ Bc) {
  int i = blockIdx.x * blockDim.x + threadIdx.x;
  if (i < N_NNZ) {
    atomicAdd(&Dc[nidx[i]], 1u);
    atomicAdd(&Bc[eidx[i]], 1u);
  }
}

// in-place u32 count -> f32 reciprocal
__global__ void invert_kernel(const unsigned* cnt, float* inv, int n) {
  int i = blockIdx.x * blockDim.x + threadIdx.x;
  if (i < n) {
    unsigned c = cnt[i];
    inv[i] = (c > 0u) ? (1.0f / (float)c) : 0.0f;
  }
}

// f32 [rows,K] row-major -> bf16 [rows,Kpad] row-major, zero padded
__global__ void conv_pad_kernel(const float* __restrict__ src, bf16_t* __restrict__ dst,
                                int rows, int K, int Kpad) {
  int i = blockIdx.x * blockDim.x + threadIdx.x;
  if (i >= rows * Kpad) return;
  int r = i / Kpad;
  int k = i - r * Kpad;
  dst[i] = (k < K) ? (bf16_t)src[r * K + k] : (bf16_t)0.0f;
}

// f32 [K,N] row-major -> bf16 [N,Kpad] row-major (transpose), zero padded
__global__ void transpose_conv_kernel(const float* __restrict__ src, bf16_t* __restrict__ dst,
                                      int K, int N, int Kpad) {
  int i = blockIdx.x * blockDim.x + threadIdx.x;
  if (i >= N * Kpad) return;
  int n = i / Kpad;
  int k = i - n * Kpad;
  dst[i] = (k < K) ? (bf16_t)src[k * N + n] : (bf16_t)0.0f;
}

// ---------------- WMMA GEMM: C[M,N] (+)= A[M,Kpad] * Bt[N,Kpad]^T ----------------
// Block = 128 threads = 4 waves stacked along M; each wave computes 32(M) x 64(N)
// (2 M-subtiles x 4 N-subtiles = 8 accumulators). The shared 64x32 bf16 B panel is
// double-buffered through LDS (conflict-free padded layout), staged via the async
// global->LDS data mover when available; A fragments stream straight from L2.
// blockIdx.z selects a K-chunk (split-K with atomic reduce).
__global__ void __launch_bounds__(128)
wmma_gemm_kernel(const bf16_t* __restrict__ A,
                 const bf16_t* __restrict__ Bt,
                 float* __restrict__ C,
                 int M, int N, int Kpad, int kChunk, int accumulate) {
  __shared__ __align__(16) bf16_t Bp[2][64 * BROW];

  const int lane = threadIdx.x & 31;
  const int wave = threadIdx.x >> 5;
  const int m0 = blockIdx.x * 128 + wave * 32;
  const int n0 = blockIdx.y * 64;
  const int kBegin = blockIdx.z * kChunk;
  const int kEnd   = kBegin + kChunk;

  const int r  = lane & 15;
  const int hi = lane >> 4;

  // cooperative B-panel staging: thread handles rows (nr, nr+32), quarter q
  const int nr = threadIdx.x >> 2;        // 0..31
  const int q  = threadIdx.x & 3;         // 0..3 (8 bf16 each)
  const bf16_t* sB0 = Bt + (n0 + nr)      * Kpad + q * 8;
  const bf16_t* sB1 = Bt + (n0 + nr + 32) * Kpad + q * 8;
  bf16_t* dB0 = &Bp[0][nr * BROW + q * 8];
  bf16_t* dB1 = &Bp[0][(nr + 32) * BROW + q * 8];
  const int bufStride = 64 * BROW;

  // A fragment pointers (ISA 16-bit A 16x32 layout: lanes<16 take K {0..7,16..23},
  // lanes>=16 take K {8..15,24..31} of each 32-wide step)
  const bf16_t* ap0 = A + (m0 + r)      * Kpad + hi * 8;
  const bf16_t* ap1 = A + (m0 + 16 + r) * Kpad + hi * 8;

  // B fragment LDS read offsets: column 16*ni + r, lanes<16 K {0..7,16..23}, lanes>=16 K {8..15,24..31}
  const int bOff = r * BROW + hi * 8;

  v8f acc[2][4];
#pragma unroll
  for (int mi = 0; mi < 2; ++mi)
#pragma unroll
    for (int ni = 0; ni < 4; ++ni) acc[mi][ni] = (v8f){};

  // prologue: stage first panel into buffer 0
#if defined(USE_ASYNC_LDS)
  async_stage_b128(sB0 + kBegin, dB0);
  async_stage_b128(sB1 + kBegin, dB1);
  __builtin_amdgcn_s_wait_asynccnt(0);
#else
  *(v8bf*)dB0 = *(const v8bf*)(sB0 + kBegin);
  *(v8bf*)dB1 = *(const v8bf*)(sB1 + kBegin);
#endif
  __syncthreads();

  int cur = 0;
  for (int k = kBegin; k < kEnd; k += 32) {
    const bool hasNext = (k + 32) < kEnd;

#if defined(USE_ASYNC_LDS)
    if (hasNext) {                         // async prefetch straight into next buffer
      async_stage_b128(sB0 + k + 32, dB0 + (cur ^ 1) * bufStride);
      async_stage_b128(sB1 + k + 32, dB1 + (cur ^ 1) * bufStride);
    }
#else
    v8bf pf0, pf1;
    if (hasNext) {                         // prefetch next panel into registers
      pf0 = *(const v8bf*)(sB0 + k + 32);
      pf1 = *(const v8bf*)(sB1 + k + 32);
    }
#endif

    // A fragments from global/L2
    v8bf a0l = *(const v8bf*)(ap0 + k);
    v8bf a0h = *(const v8bf*)(ap0 + k + 16);
    v8bf a1l = *(const v8bf*)(ap1 + k);
    v8bf a1h = *(const v8bf*)(ap1 + k + 16);
    v16bf a[2];
    a[0] = __builtin_shufflevector(a0l, a0h, 0,1,2,3,4,5,6,7,8,9,10,11,12,13,14,15);
    a[1] = __builtin_shufflevector(a1l, a1h, 0,1,2,3,4,5,6,7,8,9,10,11,12,13,14,15);

    // B fragments from LDS
    const bf16_t* base = &Bp[cur][bOff];
    v16bf b[4];
#pragma unroll
    for (int ni = 0; ni < 4; ++ni) {
      v8bf bl = *(const v8bf*)(base + ni * 16 * BROW);
      v8bf bh = *(const v8bf*)(base + ni * 16 * BROW + 16);
      b[ni] = __builtin_shufflevector(bl, bh, 0,1,2,3,4,5,6,7,8,9,10,11,12,13,14,15);
    }

#pragma unroll
    for (int mi = 0; mi < 2; ++mi)
#pragma unroll
      for (int ni = 0; ni < 4; ++ni)
        acc[mi][ni] = __builtin_amdgcn_wmma_f32_16x16x32_bf16(
            false, a[mi], false, b[ni], (short)0, acc[mi][ni], false, false);

    if (hasNext) {                         // flip buffers
#if defined(USE_ASYNC_LDS)
      __builtin_amdgcn_s_wait_asynccnt(0);
#else
      *(v8bf*)(dB0 + (cur ^ 1) * bufStride) = pf0;
      *(v8bf*)(dB1 + (cur ^ 1) * bufStride) = pf1;
#endif
      __syncthreads();
      cur ^= 1;
    }
  }

  // C/D layout: VGPR g -> row (m-subtile + hi*8 + g), col n0 + 16*ni + r
#pragma unroll
  for (int mi = 0; mi < 2; ++mi) {
    const int rowb = m0 + mi * 16 + hi * 8;
    float* cp = C + rowb * N + n0 + r;
#pragma unroll
    for (int g = 0; g < 8; ++g) {
      if (!accumulate) {
#pragma unroll
        for (int ni = 0; ni < 4; ++ni) cp[g * N + ni * 16] = acc[mi][ni][g];
      } else {
#pragma unroll
        for (int ni = 0; ni < 4; ++ni) atomicAdd(&cp[g * N + ni * 16], acc[mi][ni][g]);
      }
    }
  }
}

// ---------------- segment scatter-add: dst[sidx[i]] += src[gidx[i]] * scale[gidx[i]] ----------------
__global__ void scatter_add_kernel(const float* __restrict__ src,
                                   float* __restrict__ dst,
                                   const int* __restrict__ gidx,
                                   const int* __restrict__ sidx,
                                   const float* __restrict__ scale,
                                   int F) {
  int t = blockIdx.x * blockDim.x + threadIdx.x;
  int F4 = F >> 2;
  int total = N_NNZ * F4;
  if (t >= total) return;
  int i = t / F4;
  int f = (t - i * F4) << 2;
  int rs = gidx[i];
  int rd = sidx[i];
  float s = scale ? scale[rs] : 1.0f;
  float4 v = *(const float4*)(src + rs * F + f);
  float* d = dst + rd * F + f;
  atomicAdd(d + 0, v.x * s);
  atomicAdd(d + 1, v.y * s);
  atomicAdd(d + 2, v.z * s);
  atomicAdd(d + 3, v.w * s);
}

// feat = relu(acc * Dinv[row] + b1[col]); also emit bf16 copy for GEMM2
__global__ void finalize1_kernel(float* __restrict__ feat, bf16_t* __restrict__ featb,
                                 const float* __restrict__ Dinv, const float* __restrict__ b1) {
  int i = blockIdx.x * blockDim.x + threadIdx.x;
  if (i >= N_NODES * HIDF) return;
  int n = i >> 12;          // /4096
  int f = i & 4095;
  float v = feat[i] * Dinv[n] + b1[f];
  v = v > 0.f ? v : 0.f;
  feat[i] = v;
  featb[i] = (bf16_t)v;
}

// hid = acc * Dinv[row] + b2[col]; code = tanh(hid)
__global__ void finalize2_kernel(float* __restrict__ hid, float* __restrict__ code,
                                 const float* __restrict__ Dinv, const float* __restrict__ b2) {
  int i = blockIdx.x * blockDim.x + threadIdx.x;
  if (i >= N_NODES * CODEF) return;
  int n = i >> 6;           // /64
  int f = i & 63;
  float v = hid[i] * Dinv[n] + b2[f];
  hid[i] = v;
  code[i] = tanhf(v);
}

// ---------------- host-side orchestration ----------------

extern "C" void kernel_launch(void* const* d_in, const int* in_sizes, int n_in,
                              void* d_out, int out_size, void* d_ws, size_t ws_size,
                              hipStream_t stream) {
  const float* x    = (const float*)d_in[0];
  const int*   nidx = (const int*)  d_in[1];
  const int*   eidx = (const int*)  d_in[2];
  const float* W1   = (const float*)d_in[3];
  const float* b1   = (const float*)d_in[4];
  const float* W2   = (const float*)d_in[5];
  const float* b2   = (const float*)d_in[6];

  float* out  = (float*)d_out;
  float* feat = out;                                   // [8192, 4096]
  float* hid  = out + (size_t)N_NODES * HIDF;          // [8192, 64]
  float* code = hid + (size_t)N_NODES * CODEF;         // [8192, 64]

  // workspace carve-out (256B aligned chunks)
  char* w = (char*)d_ws;
  auto alloc = [&](size_t bytes) -> void* {
    void* p = (void*)w;
    w += (bytes + 255) & ~(size_t)255;
    return p;
  };
  float*  DB    = (float*)alloc(2 * N_NODES * sizeof(float));  // Dinv | Binv (counts first)
  float*  Dinv  = DB;
  float*  Binv  = DB + N_NODES;
  float*  xw2   = (float*)alloc((size_t)N_NODES * CODEF * sizeof(float));
  float*  m2    = (float*)alloc((size_t)N_EDGES * CODEF * sizeof(float));
  bf16_t* xbf   = (bf16_t*)alloc((size_t)N_NODES * KPAD1 * sizeof(bf16_t));
  bf16_t* w1t   = (bf16_t*)alloc((size_t)HIDF * KPAD1 * sizeof(bf16_t));
  bf16_t* w2t   = (bf16_t*)alloc((size_t)CODEF * HIDF * sizeof(bf16_t));
  bf16_t* featb = (bf16_t*)alloc((size_t)N_NODES * HIDF * sizeof(bf16_t));
  float*  m1    = (float*)alloc((size_t)N_EDGES * HIDF * sizeof(float));

  const int TB = 256;
  auto gsz = [](int n, int b) { return (n + b - 1) / b; };

  // degrees -> Dinv/Binv (integer counts, then reciprocal in place)
  zero_f32_kernel<<<gsz(2 * N_NODES / 4, TB), TB, 0, stream>>>(DB, 2 * N_NODES / 4);
  degree_count_kernel<<<gsz(N_NNZ, TB), TB, 0, stream>>>(nidx, eidx,
                                                         (unsigned*)Dinv, (unsigned*)Binv);
  invert_kernel<<<gsz(2 * N_NODES, TB), TB, 0, stream>>>((const unsigned*)DB, DB, 2 * N_NODES);

  // bf16 conversions for GEMM1
  conv_pad_kernel<<<gsz(N_NODES * KPAD1, TB), TB, 0, stream>>>(x, xbf, N_NODES, TXT_FEAT, KPAD1);
  transpose_conv_kernel<<<gsz(HIDF * KPAD1, TB), TB, 0, stream>>>(W1, w1t, TXT_FEAT, HIDF, KPAD1);

  // GEMM1: xw = x @ W1  (staged in the feat region of d_out)
  {
    dim3 grid(N_NODES / 128, HIDF / 64, 1);
    wmma_gemm_kernel<<<grid, 128, 0, stream>>>(xbf, w1t, feat,
                                               N_NODES, HIDF, KPAD1, KPAD1, 0);
  }

  // m1 = segment_sum(xw[node_idx]) over edges
  zero_f32_kernel<<<gsz(N_EDGES * HIDF / 4, TB), TB, 0, stream>>>(m1, N_EDGES * HIDF / 4);
  scatter_add_kernel<<<gsz(N_NNZ * (HIDF / 4), TB), TB, 0, stream>>>(
      feat, m1, nidx, eidx, (const float*)nullptr, HIDF);

  // feat_acc = segment_sum(m1[edge_idx] * Binv[edge_idx]) over nodes
  zero_f32_kernel<<<gsz(N_NODES * HIDF / 4, TB), TB, 0, stream>>>(feat, N_NODES * HIDF / 4);
  scatter_add_kernel<<<gsz(N_NNZ * (HIDF / 4), TB), TB, 0, stream>>>(
      m1, feat, eidx, nidx, Binv, HIDF);

  // feat = relu(feat_acc * Dinv + b1), featb = bf16(feat)
  finalize1_kernel<<<gsz(N_NODES * HIDF, TB), TB, 0, stream>>>(feat, featb, Dinv, b1);

  // GEMM2: xw2 = feat @ W2  (split-K x4, atomic reduce into zeroed buffer)
  transpose_conv_kernel<<<gsz(CODEF * HIDF, TB), TB, 0, stream>>>(W2, w2t, HIDF, CODEF, HIDF);
  zero_f32_kernel<<<gsz(N_NODES * CODEF / 4, TB), TB, 0, stream>>>(xw2, N_NODES * CODEF / 4);
  {
    dim3 grid(N_NODES / 128, CODEF / 64, 4);
    wmma_gemm_kernel<<<grid, 128, 0, stream>>>(featb, w2t, xw2,
                                               N_NODES, CODEF, HIDF, HIDF / 4, 1);
  }

  // m2 = segment_sum(xw2[node_idx]) over edges
  zero_f32_kernel<<<gsz(N_EDGES * CODEF / 4, TB), TB, 0, stream>>>(m2, N_EDGES * CODEF / 4);
  scatter_add_kernel<<<gsz(N_NNZ * (CODEF / 4), TB), TB, 0, stream>>>(
      xw2, m2, nidx, eidx, (const float*)nullptr, CODEF);

  // hid_acc = segment_sum(m2[edge_idx] * Binv[edge_idx]) over nodes
  zero_f32_kernel<<<gsz(N_NODES * CODEF / 4, TB), TB, 0, stream>>>(hid, N_NODES * CODEF / 4);
  scatter_add_kernel<<<gsz(N_NNZ * (CODEF / 4), TB), TB, 0, stream>>>(
      m2, hid, eidx, nidx, Binv, CODEF);

  // hid = hid_acc * Dinv + b2 ; code = tanh(hid)
  finalize2_kernel<<<gsz(N_NODES * CODEF, TB), TB, 0, stream>>>(hid, code, Dinv, b2);
}
